// CrossRPEAttention_9869834846448
// MI455X (gfx1250) — compile-verified, matrix operand order checked by reading, not run on hardware
//
#include <hip/hip_runtime.h>

typedef __attribute__((ext_vector_type(16))) __bf16 v16bf;
typedef __attribute__((ext_vector_type(8)))  float  v8f;

#define B_    4
#define DIM_  768
#define H_    12
#define HD_   64
#define GRID_ 24
#define P_    576
#define Q_    577
#define N_    2305
#define QPAD  592      // 37*16
#define MROWS 9232     // ceil(B*N/16)*16
#define KSTR  2336     // 73*32 padded key rows
#define MPROJ 2368     // B_*QPAD

__device__ __forceinline__ ushort f2bf(float f) {
  uint u = __float_as_uint(f);
  u += 0x7FFFu + ((u >> 16) & 1u);   // round-to-nearest-even
  return (ushort)(u >> 16);
}

__device__ __forceinline__ int koff(int j, int hi) {
  // 16-bit A/B operand K index for dword j, half-wave hi (ISA 7.12.2)
  return ((j < 4) ? 0 : 16) + hi * 8 + 2 * (j & 3);
}

__device__ __forceinline__ v8f wmma_bf16(v16bf a, v16bf b, v8f c) {
  return __builtin_amdgcn_wmma_f32_16x16x32_bf16(false, a, false, b, (short)0, c, false, false);
}

__device__ __forceinline__ float rmax16(float v) {
  #pragma unroll
  for (int o = 8; o >= 1; o >>= 1) v = fmaxf(v, __shfl_xor(v, o, 16));
  return v;
}
__device__ __forceinline__ float rsum16(float v) {
  #pragma unroll
  for (int o = 8; o >= 1; o >>= 1) v += __shfl_xor(v, o, 16);
  return v;
}

// ---------------------------------------------------------------- converts
__global__ void cvt_kernel(const float* __restrict__ src, ushort* __restrict__ dst,
                           long nsrc, long ndst) {
  long i = (long)blockIdx.x * blockDim.x + threadIdx.x;
  if (i >= ndst) return;
  dst[i] = (i < nsrc) ? f2bf(src[i]) : (ushort)0;
}

// ------------------------------------------------- q = q_learned + posembed
__global__ void qprep_kernel(const float* __restrict__ q_learned,
                             const float* __restrict__ rpe_table,
                             ushort* __restrict__ qb, float* __restrict__ tmp) {
  __shared__ float rowv[DIM_];
  const int q = blockIdx.x;          // 0..QPAD-1
  const int tid = threadIdx.x;       // 256 threads
  for (int t = tid; t < DIM_; t += 256) {
    float val = 0.f;
    if (q < Q_) {
      float pos = 0.f;
      if (q > 0) {
        int p = q - 1;
        int xj = p % GRID_, yi = p / GRID_;
        int quad = t / 192;          // 0:sin(x) 1:cos(x) 2:sin(y) 3:cos(y)
        int k = t % 192;
        double omega = pow(10000.0, -(double)k / 192.0);
        double arg = (double)((quad < 2) ? xj : yi) * omega;
        pos = (float)((quad & 1) ? cos(arg) : sin(arg));
      }
      val = q_learned[t] + pos;
    }
    rowv[t] = val;
    qb[(size_t)q * DIM_ + t] = f2bf(val);
  }
  __syncthreads();
  if (q < Q_ && tid < H_ * 6) {      // tmp[h,q,u] = q[h]·rpe_table[u]
    int h = tid / 6, u = tid % 6;
    float s = 0.f;
    for (int d = 0; d < HD_; ++d) s += rowv[h * HD_ + d] * rpe_table[u * HD_ + d];
    tmp[((size_t)h * Q_ + q) * 6 + u] = s;
  }
}

// ------------------------------------------------- rpe[h][q][kp] bias table
__global__ void rpe_kernel(const float* __restrict__ tmp, float* __restrict__ rpe) {
  long i = (long)blockIdx.x * blockDim.x + threadIdx.x;
  if (i >= (long)Q_ * Q_) return;
  int q = (int)(i / Q_), kp = (int)(i % Q_);
  int bucket;
  if (q == 0 || kp == 0) bucket = 5;                 // CLS bucket
  else {
    int a = q - 1, b = kp - 1;
    float dy = (float)(a / GRID_ - b / GRID_);
    float dx = (float)(a % GRID_ - b % GRID_);
    float dis = rintf(sqrtf(dy * dy + dx * dx));
    if (dis <= 1.9f) bucket = (int)rintf(dis);
    else {
      float far = rintf(1.9f + logf(dis / 1.9f) / logf(15.2f / 1.9f) * 1.9f);
      bucket = (int)fminf(far, 3.8f);                // trunc(3.8)=3
    }
  }
  for (int h = 0; h < H_; ++h)
    rpe[((size_t)h * Q_ + q) * Q_ + kp] = tmp[((size_t)h * Q_ + q) * 6 + bucket];
}

// -------------------------------- GEMM: C[16x64] = A[16xK] @ Bt[64xK]^T
// mode 0: K store [b][h][n][hd]   mode 1: V store transposed [b][h][hd][n]
// mode 2: output proj, f32 + bias, masked rows
__global__ __launch_bounds__(32)
void gemm_bf16_16x64(const ushort* __restrict__ A, const ushort* __restrict__ Bt,
                     int Kdim, int mode, void* __restrict__ outp,
                     const float* __restrict__ bias) {
  const int lane = threadIdx.x, row = lane & 15, hi = lane >> 4;
  const int m0 = blockIdx.x * 16, n0 = blockIdx.y * 64;
  union U { uint u[8]; v16bf v; };
  v8f acc0 = {0,0,0,0,0,0,0,0}, acc1 = acc0, acc2 = acc0, acc3 = acc0;

  for (int kk = 0; kk < Kdim; kk += 32) {
    U a, b0, b1, b2, b3;
    #pragma unroll
    for (int j = 0; j < 8; ++j) {
      int k = kk + koff(j, hi);
      a.u[j]  = *(const uint*)(A  + (size_t)(m0 + row) * Kdim + k);
      b0.u[j] = *(const uint*)(Bt + (size_t)(n0      + row) * Kdim + k);
      b1.u[j] = *(const uint*)(Bt + (size_t)(n0 + 16 + row) * Kdim + k);
      b2.u[j] = *(const uint*)(Bt + (size_t)(n0 + 32 + row) * Kdim + k);
      b3.u[j] = *(const uint*)(Bt + (size_t)(n0 + 48 + row) * Kdim + k);
    }
    acc0 = wmma_bf16(a.v, b0.v, acc0);
    acc1 = wmma_bf16(a.v, b1.v, acc1);
    acc2 = wmma_bf16(a.v, b2.v, acc2);
    acc3 = wmma_bf16(a.v, b3.v, acc3);
  }

  #pragma unroll
  for (int r = 0; r < 8; ++r) {
    int mg = m0 + r + hi * 8;
    float vals[4] = {acc0[r], acc1[r], acc2[r], acc3[r]};
    #pragma unroll
    for (int c = 0; c < 4; ++c) {
      int col = n0 + c * 16 + row;
      if (mode == 0) {
        if (mg < B_ * N_) {
          int b = mg / N_, n = mg % N_, h = col >> 6, hd = col & 63;
          ((ushort*)outp)[(((size_t)(b * H_ + h) * KSTR + n) << 6) + hd] = f2bf(vals[c]);
        }
      } else if (mode == 1) {
        if (mg < B_ * N_) {
          int b = mg / N_, n = mg % N_, h = col >> 6, hd = col & 63;
          ((ushort*)outp)[((size_t)((b * H_ + h) * HD_ + hd)) * KSTR + n] = f2bf(vals[c]);
        }
      } else {
        int b = mg / QPAD, q = mg % QPAD;
        if (b < B_ && q < Q_)
          ((float*)outp)[((size_t)(b * Q_ + q)) * DIM_ + col] = vals[c] + bias[col];
      }
    }
  }
}

// -------------------------------- flash attention: one wave per (b,h,16 q)
__global__ __launch_bounds__(32)
void attn_kernel(const ushort* __restrict__ qb, const ushort* __restrict__ Kb,
                 const ushort* __restrict__ Vt, const float* __restrict__ rpe,
                 ushort* __restrict__ aout) {
  __shared__ ushort pbuf[16 * 32];
  const int lane = threadIdx.x, row = lane & 15, hi = lane >> 4;
  const int qt = blockIdx.x, h = blockIdx.y, bz = blockIdx.z;
  const float scale = 0.125f;              // 64^-0.5
  union U { uint u[8]; v16bf v; };

  U a0, a1;                                 // q tile, d=0..31 / 32..63
  #pragma unroll
  for (int j = 0; j < 8; ++j) {
    size_t base = (size_t)(qt * 16 + row) * DIM_ + h * HD_;
    a0.u[j] = *(const uint*)(qb + base + koff(j, hi));
    a1.u[j] = *(const uint*)(qb + base + 32 + koff(j, hi));
  }

  v8f o0 = {0,0,0,0,0,0,0,0}, o1 = o0, o2 = o0, o3 = o0;
  float mr[8], lr[8];
  #pragma unroll
  for (int r = 0; r < 8; ++r) { mr[r] = -1e30f; lr[r] = 0.f; }

  const size_t kb_bh = (size_t)(bz * H_ + h) * KSTR;
  const size_t vb_bh = (size_t)(bz * H_ + h) * HD_;

  for (int kt = 0; kt < KSTR / 32; ++kt) {
    const int kb = kt * 32;
    v8f s0 = {0,0,0,0,0,0,0,0}, s1 = s0;
    {
      U bk;
      #pragma unroll
      for (int j = 0; j < 8; ++j)
        bk.u[j] = *(const uint*)(Kb + ((kb_bh + kb + row) << 6) + koff(j, hi));
      s0 = wmma_bf16(a0.v, bk.v, s0);
      #pragma unroll
      for (int j = 0; j < 8; ++j)
        bk.u[j] = *(const uint*)(Kb + ((kb_bh + kb + row) << 6) + 32 + koff(j, hi));
      s0 = wmma_bf16(a1.v, bk.v, s0);
      #pragma unroll
      for (int j = 0; j < 8; ++j)
        bk.u[j] = *(const uint*)(Kb + ((kb_bh + kb + 16 + row) << 6) + koff(j, hi));
      s1 = wmma_bf16(a0.v, bk.v, s1);
      #pragma unroll
      for (int j = 0; j < 8; ++j)
        bk.u[j] = *(const uint*)(Kb + ((kb_bh + kb + 16 + row) << 6) + 32 + koff(j, hi));
      s1 = wmma_bf16(a1.v, bk.v, s1);
    }

    // online softmax, per C-matrix row (16 lanes own one row)
    #pragma unroll
    for (int r = 0; r < 8; ++r) {
      int m = r + hi * 8;
      int qg = qt * 16 + m; int qc = qg > 576 ? 576 : qg;
      int k0 = kb + row, k1 = kb + 16 + row;
      float l0 = -1e30f, l1 = -1e30f;
      if (k0 < N_) {
        int kp = (k0 == 0) ? 0 : 1 + (k0 - 1) % P_;
        l0 = s0[r] * scale + rpe[((size_t)h * Q_ + qc) * Q_ + kp];
      }
      if (k1 < N_) {
        int kp = (k1 == 0) ? 0 : 1 + (k1 - 1) % P_;
        l1 = s1[r] * scale + rpe[((size_t)h * Q_ + qc) * Q_ + kp];
      }
      float mn = fmaxf(mr[r], rmax16(fmaxf(l0, l1)));
      float corr = __expf(mr[r] - mn);
      float p0 = __expf(l0 - mn), p1 = __expf(l1 - mn);
      lr[r] = lr[r] * corr + rsum16(p0 + p1);
      mr[r] = mn;
      o0[r] *= corr; o1[r] *= corr; o2[r] *= corr; o3[r] *= corr;
      pbuf[m * 32 + row]      = f2bf(p0);
      pbuf[m * 32 + 16 + row] = f2bf(p1);
    }
    __asm__ volatile("s_wait_dscnt 0" ::: "memory");

    U pa;                                    // P tile reloaded in A layout
    #pragma unroll
    for (int j = 0; j < 8; ++j)
      pa.u[j] = *(const uint*)(pbuf + row * 32 + koff(j, hi));

    U bv;
    #pragma unroll
    for (int c = 0; c < 4; ++c) {
      #pragma unroll
      for (int j = 0; j < 8; ++j)
        bv.u[j] = *(const uint*)(Vt + (vb_bh + c * 16 + row) * KSTR + kb + koff(j, hi));
      if      (c == 0) o0 = wmma_bf16(pa.v, bv.v, o0);
      else if (c == 1) o1 = wmma_bf16(pa.v, bv.v, o1);
      else if (c == 2) o2 = wmma_bf16(pa.v, bv.v, o2);
      else             o3 = wmma_bf16(pa.v, bv.v, o3);
    }
    __asm__ volatile("s_wait_dscnt 0" ::: "memory");  // WAR guard on pbuf
  }

  #pragma unroll
  for (int r = 0; r < 8; ++r) {
    int m = r + hi * 8;
    float inv = 1.0f / lr[r];
    size_t ob = ((size_t)(bz * QPAD + qt * 16 + m)) * DIM_ + h * HD_ + row;
    aout[ob]      = f2bf(o0[r] * inv);
    aout[ob + 16] = f2bf(o1[r] * inv);
    aout[ob + 32] = f2bf(o2[r] * inv);
    aout[ob + 48] = f2bf(o3[r] * inv);
  }
}

// ---------------------------------------------------------------- launcher
extern "C" void kernel_launch(void* const* d_in, const int* in_sizes, int n_in,
                              void* d_out, int out_size, void* d_ws, size_t ws_size,
                              hipStream_t stream) {
  const float* x         = (const float*)d_in[0];
  const float* wk        = (const float*)d_in[1];
  const float* wv        = (const float*)d_in[2];
  const float* proj_w    = (const float*)d_in[3];
  const float* proj_b    = (const float*)d_in[4];
  const float* q_learned = (const float*)d_in[5];
  const float* rpe_table = (const float*)d_in[6];

  char* ws = (char*)d_ws;
  size_t off = 0;
  auto carve = [&](size_t bytes) -> char* {
    char* p = ws + off;
    off = (off + bytes + 255) & ~(size_t)255;
    return p;
  };
  ushort* xb   = (ushort*)carve((size_t)MROWS * DIM_ * 2);
  ushort* wkb  = (ushort*)carve((size_t)DIM_ * DIM_ * 2);
  ushort* wvb  = (ushort*)carve((size_t)DIM_ * DIM_ * 2);
  ushort* pwb  = (ushort*)carve((size_t)DIM_ * DIM_ * 2);
  ushort* qb   = (ushort*)carve((size_t)QPAD * DIM_ * 2);
  float*  tmp  = (float*) carve((size_t)H_ * Q_ * 6 * 4);
  float*  rpe  = (float*) carve((size_t)H_ * Q_ * Q_ * 4);
  ushort* Kb   = (ushort*)carve((size_t)B_ * H_ * KSTR * HD_ * 2);
  ushort* Vt   = (ushort*)carve((size_t)B_ * H_ * KSTR * HD_ * 2);
  ushort* aout = (ushort*)carve((size_t)MPROJ * DIM_ * 2);

  long nx  = (long)B_ * N_ * DIM_;
  long nxp = (long)MROWS * DIM_;
  long nw  = (long)DIM_ * DIM_;
  cvt_kernel<<<(int)((nxp + 255) / 256), 256, 0, stream>>>(x, xb, nx, nxp);
  cvt_kernel<<<(int)((nw + 255) / 256), 256, 0, stream>>>(wk, wkb, nw, nw);
  cvt_kernel<<<(int)((nw + 255) / 256), 256, 0, stream>>>(wv, wvb, nw, nw);
  cvt_kernel<<<(int)((nw + 255) / 256), 256, 0, stream>>>(proj_w, pwb, nw, nw);

  qprep_kernel<<<QPAD, 256, 0, stream>>>(q_learned, rpe_table, qb, tmp);
  long nr = (long)Q_ * Q_;
  rpe_kernel<<<(int)((nr + 255) / 256), 256, 0, stream>>>(tmp, rpe);

  dim3 gkv(MROWS / 16, DIM_ / 64);
  gemm_bf16_16x64<<<gkv, 32, 0, stream>>>(xb, wkb, DIM_, 0, Kb, nullptr);
  gemm_bf16_16x64<<<gkv, 32, 0, stream>>>(xb, wvb, DIM_, 1, Vt, nullptr);

  dim3 ga(QPAD / 16, H_, B_);
  attn_kernel<<<ga, 32, 0, stream>>>(qb, Kb, Vt, rpe, aout);

  dim3 gp(MPROJ / 16, DIM_ / 64);
  gemm_bf16_16x64<<<gp, 32, 0, stream>>>(aout, pwb, DIM_, 2, d_out, proj_b);
}
